// CustomRNN_34368328303082
// MI455X (gfx1250) — matrix operand order repaired and checked by reading, not compiled, not run on hardware
//
#include <hip/hip_runtime.h>
#include <hip/hip_bf16.h>

// ---------------------------------------------------------------------------
// CDNA5 (gfx1250) 2-layer LSTM with mask/reset.
//   T=2048, B=64, I=H=256, gates=4H=1024.
//   1) xg = x @ W_ih^T + b : parallel WMMA bf16 GEMM (compute bound, ~69 GF/layer)
//   2) sequential scan: gates = xg_t + h @ W_hh^T, LSTM cell, mask/reset.
//      4 WGs (16 batch rows = one WMMA M-tile each), h/c state in regs+LDS,
//      W_hh (bf16, 512KB) streamed from the 192MB L2, xg tile double-buffered
//      in LDS via CDNA5 async global->LDS loads (ASYNCcnt).
// ---------------------------------------------------------------------------

#define TT 2048
#define BB 64
#define II 256
#define HH 256
#define GG 1024  // 4*H

typedef __attribute__((ext_vector_type(16))) __bf16 v16bf;
typedef __attribute__((ext_vector_type(8)))  __bf16 v8bf;
typedef __attribute__((ext_vector_type(8)))  float  v8f;

#if __has_builtin(__builtin_amdgcn_global_load_async_to_lds_b128) && \
    __has_builtin(__builtin_amdgcn_s_wait_asynccnt)
#define HAS_ASYNC 1
// exact payload type from the builtin's signature: v4i in AS1 (global) / AS3 (LDS)
typedef __attribute__((vector_size(16))) int v4i_;
typedef __attribute__((address_space(1))) v4i_ g_v4i;
typedef __attribute__((address_space(3))) v4i_ l_v4i;
#else
#define HAS_ASYNC 0
#endif

__device__ __forceinline__ v16bf cat16(v8bf lo, v8bf hi) {
    return __builtin_shufflevector(lo, hi, 0,1,2,3,4,5,6,7,8,9,10,11,12,13,14,15);
}

__device__ __forceinline__ float sigmoidf_(float x) {
    return 1.0f / (1.0f + __expf(-x));
}

// ---------------------------------------------------------------------------
// f32 -> bf16 conversion (weights)
// ---------------------------------------------------------------------------
__global__ void cvt_f32_bf16(const float* __restrict__ in, __bf16* __restrict__ out, int n) {
    int i = blockIdx.x * blockDim.x + threadIdx.x;
    if (i < n) out[i] = (__bf16)in[i];
}

// ---------------------------------------------------------------------------
// xg GEMM: XG[M,1024] = X[M,256] @ W[1024,256]^T + (b_ih + b_hh)
// One wave -> 16x64 output tile (4 N-tiles), K-loop of 8.
// A frag: lane(l) row = m0+(l&15), K = k0 + (l>>4)*16 + {0..15} (contiguous)
// B frag: lane(l) col = n0+(l&15), K = k0 + (l>>4)*16 + {0..15} (contiguous)
// C/D   : lane(l) col N = l&15, vgpr r -> row M = r + (l>>4)*8
// ---------------------------------------------------------------------------
__global__ void __launch_bounds__(256)
xg_gemm(const float* __restrict__ X, const __bf16* __restrict__ W,
        const float* __restrict__ b_ih, const float* __restrict__ b_hh,
        float* __restrict__ XG) {
    const int lane   = threadIdx.x & 31;
    const int wv     = threadIdx.x >> 5;
    const int task   = blockIdx.x * 8 + wv;
    const int mt     = task >> 4;
    const int ng     = task & 15;
    const int m0     = mt * 16;
    const int lanelo = lane & 15;
    const int hk8    = (lane >> 4) * 8;
    const int hk16   = (lane >> 4) * 16;

    v8f acc[4] = {};

    #pragma unroll
    for (int kt = 0; kt < 8; ++kt) {
        const int k0 = kt * 32;
        const float* xrow = X + (size_t)(m0 + lanelo) * II + k0 + hk16;
        v8f  xlo = *(const v8f*)xrow;
        v8f  xhi = *(const v8f*)(xrow + 8);
        v8bf alo = __builtin_convertvector(xlo, v8bf);
        v8bf ahi = __builtin_convertvector(xhi, v8bf);
        v16bf a  = cat16(alo, ahi);
        #pragma unroll
        for (int i = 0; i < 4; ++i) {
            const int n0 = ng * 64 + i * 16;
            v16bf b = *(const v16bf*)(W + (size_t)(n0 + lanelo) * II + k0 + hk16);
            acc[i] = __builtin_amdgcn_wmma_f32_16x16x32_bf16(
                false, a, false, b, (short)0, acc[i], false, false);
        }
    }

    #pragma unroll
    for (int i = 0; i < 4; ++i) {
        const int n    = ng * 64 + i * 16 + lanelo;
        const float bv = b_ih[n] + b_hh[n];
        #pragma unroll
        for (int r = 0; r < 8; ++r) {
            XG[(size_t)(m0 + r + hk8) * GG + n] = acc[i][r] + bv;
        }
    }
}

// ---------------------------------------------------------------------------
// Recurrent scan, one layer. Grid = 4 WGs x 512 threads (16 wave32s).
// Per step (per WG):  gates(16x1024) = xg_t + h(16x256) @ Whh(1024x256)^T
//   - xg tile (64KB) is async-loaded into an LDS double buffer one step ahead
//   - each wave: 4 N-tiles x 8 K-tiles = 32 v_wmma_f32_16x16x32_bf16
//   - gates overwrite the consumed xg buffer (same lane reads then writes)
//   - c and f32-h carries live in per-thread registers; only bf16 h (the
//     next step's WMMA A operand) goes through LDS.
// ---------------------------------------------------------------------------
__global__ void __launch_bounds__(512)
lstm_scan(const float* __restrict__ XG, const float* __restrict__ mask,
          const float* __restrict__ reset, const __bf16* __restrict__ Whh,
          float* __restrict__ out, float* __restrict__ hn, float* __restrict__ cn) {
#if HAS_ASYNC
    __shared__ __align__(32) float  lds_xg[2][16 * GG];   // 128 KB double buffer
#else
    __shared__ __align__(32) float  lds_gates[16 * GG];   // 64 KB
#endif
    __shared__ __align__(32) __bf16 lds_h[16 * HH];       //  8 KB bf16 A operand

    const int tid    = threadIdx.x;
    const int lane   = tid & 31;
    const int wv     = tid >> 5;            // 0..15
    const int b0     = blockIdx.x * 16;     // batch slice
    const int lanelo = lane & 15;
    const int hk8    = (lane >> 4) * 8;
    const int hk16   = (lane >> 4) * 16;

    // per-thread LSTM state: element idx = tid + e*512 -> (b = idx>>8, j = idx&255)
    float creg[8], hreg[8];
    #pragma unroll
    for (int e = 0; e < 8; ++e) { creg[e] = 0.0f; hreg[e] = 0.0f; }

    for (int idx = tid; idx < 16 * HH; idx += 512)
        lds_h[idx] = (__bf16)0.0f;

#if HAS_ASYNC
    // preload xg tile for t=0
    {
        char* g0 = (char*)(XG + (size_t)b0 * GG);
        char* l0 = (char*)&lds_xg[0][0];
        #pragma unroll
        for (int k = 0; k < 8; ++k) {
            const int off = (tid + k * 512) * 16;
            __builtin_amdgcn_global_load_async_to_lds_b128(
                (g_v4i*)(g0 + off), (l_v4i*)(l0 + off), 0, 0);
        }
        __builtin_amdgcn_s_wait_asynccnt(0);
    }
#endif
    __syncthreads();

    for (int t = 0; t < TT; ++t) {
        const size_t rowbase = (size_t)t * BB + b0;
#if HAS_ASYNC
        float* gbuf = &lds_xg[t & 1][0];
        // kick off async load of next step's xg tile into the other buffer
        if (t + 1 < TT) {
            char* gsrc = (char*)(XG + (rowbase + BB) * GG);
            char* ldst = (char*)&lds_xg[(t + 1) & 1][0];
            #pragma unroll
            for (int k = 0; k < 8; ++k) {
                const int off = (tid + k * 512) * 16;
                __builtin_amdgcn_global_load_async_to_lds_b128(
                    (g_v4i*)(gsrc + off), (l_v4i*)(ldst + off), 0, 0);
            }
        }
#else
        float* gbuf = lds_gates;
        if (t + 1 < TT)
            __builtin_prefetch((const char*)(XG + (rowbase + BB) * GG) + tid * 128, 0, 0);
#endif

        // ---- WMMA phase: load the 8 A fragments of h once, reuse x4 tiles --
        v16bf afrag[8];
        #pragma unroll
        for (int kt = 0; kt < 8; ++kt) {
            const __bf16* hp = &lds_h[lanelo * HH + kt * 32 + hk16];
            v8bf lo = *(const v8bf*)hp;
            v8bf hi = *(const v8bf*)(hp + 8);
            afrag[kt] = cat16(lo, hi);
        }
        #pragma unroll
        for (int i = 0; i < 4; ++i) {
            const int n0   = (wv * 4 + i) * 16;
            const int ncol = n0 + lanelo;
            v8f acc;
            #pragma unroll
            for (int r = 0; r < 8; ++r)
#if HAS_ASYNC
                acc[r] = gbuf[(r + hk8) * GG + ncol];               // LDS (prefetched)
#else
                acc[r] = XG[(rowbase + r + hk8) * GG + ncol];       // global
#endif
            #pragma unroll
            for (int kt = 0; kt < 8; ++kt) {
                v16bf b = *(const v16bf*)(Whh + (size_t)(n0 + lanelo) * HH + kt * 32 + hk16);
                acc = __builtin_amdgcn_wmma_f32_16x16x32_bf16(
                    false, afrag[kt], false, b, (short)0, acc, false, false);
            }
            // gates overwrite the xg slots this lane just consumed
            #pragma unroll
            for (int r = 0; r < 8; ++r)
                gbuf[(r + hk8) * GG + ncol] = acc[r];
        }
        __syncthreads();

        // ---- elementwise LSTM cell: 4096 elems / 512 threads ---------------
        #pragma unroll
        for (int e = 0; e < 8; ++e) {
            const int idx = tid + e * 512;
            const int b   = idx >> 8;        // 0..15
            const int j   = idx & 255;       // 0..255
            const float ig = gbuf[b * GG + j];
            const float fg = gbuf[b * GG + 256 + j];
            const float gg = gbuf[b * GG + 512 + j];
            const float og = gbuf[b * GG + 768 + j];
            const float c_old = creg[e];
            const float h_old = hreg[e];
            const float c_new = sigmoidf_(fg) * c_old + sigmoidf_(ig) * tanhf(gg);
            const float h_new = sigmoidf_(og) * tanhf(c_new);
            const float m  = mask[(size_t)t * BB + b0 + b];
            const float rr = reset[(size_t)t * BB + b0 + b];
            const float h_b = h_new * m + h_old * (1.0f - m);
            const float c_b = c_new * m + c_old * (1.0f - m);
            out[((size_t)t * BB + b0 + b) * HH + j] = h_b;
            const float h_c = h_b * (1.0f - rr);
            const float c_c = c_b * (1.0f - rr);
            hreg[e] = h_c;
            creg[e] = c_c;
            lds_h[b * HH + j] = (__bf16)h_c;
        }
#if HAS_ASYNC
        __builtin_amdgcn_s_wait_asynccnt(0);   // own async loads landed
#endif
        __syncthreads();                       // all waves' loads + EW visible
    }

    // ---- final carries -----------------------------------------------------
    #pragma unroll
    for (int e = 0; e < 8; ++e) {
        const int idx = tid + e * 512;
        const int b = idx >> 8, j = idx & 255;
        hn[(size_t)(b0 + b) * HH + j] = hreg[e];
        cn[(size_t)(b0 + b) * HH + j] = creg[e];
    }
}

// ---------------------------------------------------------------------------
// Host-side orchestration
// ---------------------------------------------------------------------------
extern "C" void kernel_launch(void* const* d_in, const int* in_sizes, int n_in,
                              void* d_out, int out_size, void* d_ws, size_t ws_size,
                              hipStream_t stream) {
    const float* input_ = (const float*)d_in[0];   // (T,B,I)
    const float* mask   = (const float*)d_in[1];   // (T,B)
    const float* reset  = (const float*)d_in[2];   // (T,B)
    const float* W_ih0  = (const float*)d_in[3];   // (1024,256)
    const float* W_hh0  = (const float*)d_in[4];
    const float* b_ih0  = (const float*)d_in[5];
    const float* b_hh0  = (const float*)d_in[6];
    const float* W_ih1  = (const float*)d_in[7];
    const float* W_hh1  = (const float*)d_in[8];
    const float* b_ih1  = (const float*)d_in[9];
    const float* b_hh1  = (const float*)d_in[10];

    const size_t M  = (size_t)TT * BB;          // 131072
    const size_t NW = (size_t)GG * HH;          // 262144 weight elements

    // workspace layout (256B aligned)
    char* ws = (char*)d_ws;
    float*  xg    = (float*)(ws);                                   // M*1024 f32 = 512MB
    float*  out0  = (float*)(ws + M * GG * sizeof(float));          // M*256 f32  = 128MB
    __bf16* wih0b = (__bf16*)(ws + M * GG * sizeof(float) + M * HH * sizeof(float));
    __bf16* whh0b = wih0b + NW;
    __bf16* wih1b = whh0b + NW;
    __bf16* whh1b = wih1b + NW;

    float* out1 = (float*)d_out;                          // (T,B,H)
    float* hn   = out1 + M * HH;                          // (2,B,H)
    float* cn   = hn + 2 * (size_t)BB * HH;               // (2,B,H)

    const int cvt_blk = 256;
    const int cvt_grd = (int)((NW + cvt_blk - 1) / cvt_blk);
    cvt_f32_bf16<<<cvt_grd, cvt_blk, 0, stream>>>(W_ih0, wih0b, (int)NW);
    cvt_f32_bf16<<<cvt_grd, cvt_blk, 0, stream>>>(W_hh0, whh0b, (int)NW);
    cvt_f32_bf16<<<cvt_grd, cvt_blk, 0, stream>>>(W_ih1, wih1b, (int)NW);
    cvt_f32_bf16<<<cvt_grd, cvt_blk, 0, stream>>>(W_hh1, whh1b, (int)NW);

    // layer 0
    xg_gemm<<<16384, 256, 0, stream>>>(input_, wih0b, b_ih0, b_hh0, xg);
    lstm_scan<<<4, 512, 0, stream>>>(xg, mask, reset, whh0b, out0, hn, cn);
    // layer 1
    xg_gemm<<<16384, 256, 0, stream>>>(out0, wih1b, b_ih1, b_hh1, xg);
    lstm_scan<<<4, 512, 0, stream>>>(xg, mask, reset, whh1b, out1,
                                     hn + (size_t)BB * HH, cn + (size_t)BB * HH);
}
